// Block_28845000360159
// MI455X (gfx1250) — matrix operand order repaired and checked by reading, not compiled, not run on hardware
//
#include <hip/hip_runtime.h>
#include <hip/hip_bf16.h>

typedef __attribute__((ext_vector_type(16))) _Float16 v16h;
typedef __attribute__((ext_vector_type(8)))  _Float16 v8h;
typedef __attribute__((ext_vector_type(8)))  float    v8f;

#define BB 128
#define TT 256
#define CC 384
#define HH 6
#define HS 64
#define FF 1536
#define MM (BB*TT)   // 32768 tokens

union FragU { v16h v; v8h h[2]; };

__device__ __forceinline__ v8f wmma16(v16h a, v16h b, v8f c) {
  // D = A(16x32 f16) * B(32x16 f16) + C(16x16 f32)
  return __builtin_amdgcn_wmma_f32_16x16x32_f16(false, a, false, b, (short)0, c,
                                                false, false);
}

// A fragment: A row-major MxK (lda), 16x32 tile at (m0,k0).
// Lane layout (ISA 7.12.2): row M = lane&15; halves 0-7 -> K = grp*8+h,
// halves 8-15 -> K = 16+grp*8+(h-8), grp = lane>>4.  Two contiguous b128 loads.
__device__ __forceinline__ v16h ldA(const _Float16* __restrict__ A, int lda,
                                    int m0, int k0, int lane) {
  int grp = lane >> 4;
  const _Float16* p = A + (size_t)(m0 + (lane & 15)) * lda + k0;
  FragU u;
  u.h[0] = *(const v8h*)(p + grp * 8);
  u.h[1] = *(const v8h*)(p + 16 + grp * 8);
  return u.v;
}

// B fragment from Bt = B^T stored row-major NxK (ldb), 32x16 tile at (k0,n0).
// Lane layout: col N = lane&15; halves 0-7 -> K = grp*8+h, halves 8-15 -> K = 16+grp*8+(h-8).
__device__ __forceinline__ v16h ldB(const _Float16* __restrict__ Bt, int ldb,
                                    int k0, int n0, int lane) {
  int grp = lane >> 4;
  const _Float16* p = Bt + (size_t)(n0 + (lane & 15)) * ldb + k0;
  FragU u;
  u.h[0] = *(const v8h*)(p + grp * 8);
  u.h[1] = *(const v8h*)(p + 16 + grp * 8);
  return u.v;
}

// Generic WMMA GEMM: C[M,N] = A[M,K] @ B[K,N]  (B given transposed, NxK)
// Block = 256 thr = 8 waves; block tile 128x64; wave tile 32x32 (2x2 WMMA).
// mode 0: f16 row-major out
// mode 1: f16 out transposed per head -> [b][h][d][t]   (for V)
// mode 2: f32 out = resid + acc + bias[n]
// mode 3: f16 out = relu(acc + bias[n])
__global__ __launch_bounds__(256) void gemm_wmma(
    const _Float16* __restrict__ A, const _Float16* __restrict__ Bt,
    const float* __restrict__ bias, const float* __restrict__ resid,
    void* __restrict__ out, int N, int K, int mode) {
  int lane = threadIdx.x & 31, wid = threadIdx.x >> 5;
  int m0 = blockIdx.y * 128 + (wid >> 1) * 32;
  int n0 = blockIdx.x * 64 + (wid & 1) * 32;
  v8f acc[2][2] = {};
  for (int k0 = 0; k0 < K; k0 += 32) {
    // prefetch the A stream two K-chunks ahead (global_prefetch_b8)
    if (k0 + 64 < K) {
      __builtin_prefetch(A + (size_t)(m0 + (lane & 15)) * K + k0 + 64, 0, 3);
      __builtin_prefetch(A + (size_t)(m0 + 16 + (lane & 15)) * K + k0 + 64, 0, 3);
    }
    v16h a0 = ldA(A, K, m0, k0, lane);
    v16h a1 = ldA(A, K, m0 + 16, k0, lane);
    v16h b0 = ldB(Bt, K, k0, n0, lane);
    v16h b1 = ldB(Bt, K, k0, n0 + 16, lane);
    acc[0][0] = wmma16(a0, b0, acc[0][0]);
    acc[0][1] = wmma16(a0, b1, acc[0][1]);
    acc[1][0] = wmma16(a1, b0, acc[1][0]);
    acc[1][1] = wmma16(a1, b1, acc[1][1]);
  }
  int grp = lane >> 4, nl = lane & 15;
  for (int i = 0; i < 2; ++i)
    for (int j = 0; j < 2; ++j)
      for (int v = 0; v < 8; ++v) {
        int m = m0 + i * 16 + grp * 8 + v;   // D layout: M = grp*8 + vgpr
        int n = n0 + j * 16 + nl;            //           N = lane&15
        float val = acc[i][j][v];
        if (mode == 0) {
          ((_Float16*)out)[(size_t)m * N + n] = (_Float16)val;
        } else if (mode == 1) {
          // m = b*256+t ; n = h*64+d  ->  vT[((b*H+h)*64+d)*256 + t]
          ((_Float16*)out)[(((size_t)(m >> 8) * HH + (n >> 6)) * HS + (n & 63)) * TT
                           + (m & 255)] = (_Float16)val;
        } else if (mode == 2) {
          size_t idx = (size_t)m * N + n;
          ((float*)out)[idx] = resid[idx] + val + bias[n];
        } else {
          float r = val + bias[n];
          ((_Float16*)out)[(size_t)m * N + n] = (_Float16)(r > 0.f ? r : 0.f);
        }
      }
}

// LayerNorm over C=384, one row per block (128 threads x 3 elems).
__global__ __launch_bounds__(128) void ln_kernel(const float* __restrict__ x,
    const float* __restrict__ g, const float* __restrict__ bta,
    _Float16* __restrict__ out) {
  __shared__ float red[128];
  int row = blockIdx.x, tid = threadIdx.x;
  const float* xr = x + (size_t)row * CC;
  float v0 = xr[tid], v1 = xr[tid + 128], v2 = xr[tid + 256];
  red[tid] = v0 + v1 + v2;
  __syncthreads();
  for (int o = 64; o > 0; o >>= 1) { if (tid < o) red[tid] += red[tid + o]; __syncthreads(); }
  float mean = red[0] * (1.0f / CC);
  __syncthreads();
  float d0 = v0 - mean, d1 = v1 - mean, d2 = v2 - mean;
  red[tid] = d0 * d0 + d1 * d1 + d2 * d2;
  __syncthreads();
  for (int o = 64; o > 0; o >>= 1) { if (tid < o) red[tid] += red[tid + o]; __syncthreads(); }
  float rstd = rsqrtf(red[0] * (1.0f / CC) + 1e-5f);
  _Float16* orow = out + (size_t)row * CC;
  orow[tid]       = (_Float16)(d0 * rstd * g[tid]       + bta[tid]);
  orow[tid + 128] = (_Float16)(d1 * rstd * g[tid + 128] + bta[tid + 128]);
  orow[tid + 256] = (_Float16)(d2 * rstd * g[tid + 256] + bta[tid + 256]);
}

// Attention: one workgroup per (b,h).
// LDS: sP = 256x256 f16 scores (128 KB) + sV = 64x256 f16 V-tile (32 KB) = 160 KB
// -> exactly 2 workgroups per 320 KB WGP.
// The V tile is copied global->LDS with CDNA5 async-to-LDS ops issued at kernel
// entry (ASYNCcnt), overlapping the whole QK^T + softmax phases; fenced with
// s_wait_asynccnt before the P@V phase.
__global__ __launch_bounds__(256) void attn_kernel(
    const _Float16* __restrict__ q, const _Float16* __restrict__ k,
    const _Float16* __restrict__ vT, _Float16* __restrict__ attn) {
  extern __shared__ _Float16 smem[];
  _Float16* sP = smem;                 // 65536 halves
  _Float16* sV = smem + (size_t)TT * TT;  // 16384 halves: [d][t], ldb=TT
  int bh = blockIdx.x, b = bh / HH, h = bh % HH;
  int lane = threadIdx.x & 31, wid = threadIdx.x >> 5;
  const _Float16* qb = q + (size_t)b * TT * CC + h * HS;  // A: rows t, lda=CC
  const _Float16* kb = k + (size_t)b * TT * CC + h * HS;  // Bt: rows s, ldb=CC
  const _Float16* vb = vT + (size_t)bh * HS * TT;         // [d][t] in global
  const float scale = 0.05103103630798288f;               // 384^-0.5 (per ref!)
  const _Float16 NEG_INF = (_Float16)(-__builtin_inff());

  // ---- kick off async V-tile copy: 16384 halves = 2048 x b128 ----
  {
    int tid = threadIdx.x;
#pragma unroll
    for (int r = 0; r < 8; ++r) {
      int idx = (r * 256 + tid) * 8;                       // half offset, 16B chunks
      unsigned dlds = (unsigned)(size_t)(sV + idx);        // LDS addr = low 32 bits
      const _Float16* gsrc = vb + idx;
      asm volatile("global_load_async_to_lds_b128 %0, %1, off"
                   :: "v"(dlds), "v"(gsrc) : "memory");
    }
  }

  // ---- phase 1: S = Q @ K^T, scale, causal mask (keep s<=t) -> LDS f16 ----
  for (int tr = wid * 2; tr < wid * 2 + 2; ++tr) {
    for (int sc = 0; sc < 16; ++sc) {
      v8f acc = {};
      acc = wmma16(ldA(qb, CC, tr * 16, 0,  lane), ldB(kb, CC, 0,  sc * 16, lane), acc);
      acc = wmma16(ldA(qb, CC, tr * 16, 32, lane), ldB(kb, CC, 32, sc * 16, lane), acc);
      int s  = sc * 16 + (lane & 15);
      int tb = tr * 16 + (lane >> 4) * 8;
      for (int v = 0; v < 8; ++v) {
        int t = tb + v;
        sP[t * TT + s] = (s <= t) ? (_Float16)(acc[v] * scale) : NEG_INF;
      }
    }
  }
  __syncthreads();

  // ---- phase 2: softmax over t (query axis) for column s = threadIdx.x ----
  {
    int s = threadIdx.x;
    float m = -__builtin_inff();
    for (int t = 0; t < TT; ++t) { float f = (float)sP[t * TT + s]; m = f > m ? f : m; }
    float sum = 0.f;
    for (int t = 0; t < TT; ++t) {
      float e = __expf((float)sP[t * TT + s] - m);  // exp(-inf)=0 kills mask
      sum += e;
      sP[t * TT + s] = (_Float16)e;
    }
    float r = 1.0f / sum;
    for (int t = 0; t < TT; ++t) sP[t * TT + s] = (_Float16)((float)sP[t * TT + s] * r);
  }
  // V tile must be resident in LDS before any wave enters phase 3.
  asm volatile("s_wait_asynccnt 0x0" ::: "memory");
  __syncthreads();

  // ---- phase 3: attn = P @ V   (A from sP, B from sV, both LDS) ----
  for (int tr = wid * 2; tr < wid * 2 + 2; ++tr) {
    v8f acc[4] = {};
    for (int s0 = 0; s0 < TT; s0 += 32) {
      int grp = lane >> 4;
      const _Float16* p = &sP[(size_t)(tr * 16 + (lane & 15)) * TT + s0];
      FragU u;
      u.h[0] = *(const v8h*)(p + grp * 8);        // ds_load_b128
      u.h[1] = *(const v8h*)(p + 16 + grp * 8);
      for (int dc = 0; dc < 4; ++dc)
        acc[dc] = wmma16(u.v, ldB(sV, TT, s0, dc * 16, lane), acc[dc]);
    }
    int grp = lane >> 4, nl = lane & 15;
    for (int dc = 0; dc < 4; ++dc)
      for (int v = 0; v < 8; ++v) {
        int t = tr * 16 + grp * 8 + v;
        int d = dc * 16 + nl;
        attn[((size_t)b * TT + t) * CC + h * HS + d] = (_Float16)acc[dc][v];
      }
  }
}

// wq/wk/wv (H,C,HS) -> W^T f16 as (N=C out, K=C): out[j*C + c] = w[h][c][d], j=h*64+d
__global__ void conv_qkv(const float* __restrict__ w, _Float16* __restrict__ out) {
  int idx = blockIdx.x * blockDim.x + threadIdx.x;
  if (idx >= CC * CC) return;
  int j = idx / CC, c = idx % CC;
  out[idx] = (_Float16)w[(size_t)(j >> 6) * CC * HS + (size_t)c * HS + (j & 63)];
}

// generic f32 (K,N) -> f16 transposed (N,K)
__global__ void conv_T(const float* __restrict__ w, _Float16* __restrict__ out,
                       int K, int N) {
  int idx = blockIdx.x * blockDim.x + threadIdx.x;
  if (idx >= K * N) return;
  int n = idx / K, kk = idx % K;
  out[(size_t)n * K + kk] = (_Float16)w[(size_t)kk * N + n];
}

extern "C" void kernel_launch(void* const* d_in, const int* in_sizes, int n_in,
                              void* d_out, int out_size, void* d_ws, size_t ws_size,
                              hipStream_t stream) {
  const float* x      = (const float*)d_in[0];
  const float* wq     = (const float*)d_in[1];
  const float* wk     = (const float*)d_in[2];
  const float* wv     = (const float*)d_in[3];
  const float* w_proj = (const float*)d_in[4];
  const float* b_proj = (const float*)d_in[5];
  const float* w1     = (const float*)d_in[6];
  const float* b1     = (const float*)d_in[7];
  const float* w2     = (const float*)d_in[8];
  const float* b2     = (const float*)d_in[9];
  const float* ln1_g  = (const float*)d_in[10];
  const float* ln1_b  = (const float*)d_in[11];
  const float* ln2_g  = (const float*)d_in[12];
  const float* ln2_b  = (const float*)d_in[13];

  // ---- workspace layout (~180 MB, with reuse) ----
  _Float16* h1   = (_Float16*)d_ws;                    // 25 MB
  _Float16* qf   = h1  + (size_t)MM * CC;              // 25 MB
  _Float16* kf   = qf  + (size_t)MM * CC;              // 25 MB
  _Float16* vTf  = kf  + (size_t)MM * CC;              // 25 MB
  _Float16* attn = vTf + (size_t)MM * CC;              // 25 MB
  _Float16* relu = h1;    // reuse h1+q+k+vT (exactly MM*FF halves) after attn
  _Float16* h2   = attn;  // reuse attn region after proj
  float* x1 = (float*)((char*)d_ws + (size_t)5 * MM * CC * sizeof(_Float16)); // 50 MB
  _Float16* WqT = (_Float16*)((char*)x1 + (size_t)MM * CC * sizeof(float));
  _Float16* WkT = WqT + (size_t)CC * CC;
  _Float16* WvT = WkT + (size_t)CC * CC;
  _Float16* WpT = WvT + (size_t)CC * CC;
  _Float16* W1T = WpT + (size_t)CC * CC;
  _Float16* W2T = W1T + (size_t)CC * FF;

  // weight conversion (f32 -> f16, transposed to NxK)
  conv_qkv<<<(CC * CC + 255) / 256, 256, 0, stream>>>(wq, WqT);
  conv_qkv<<<(CC * CC + 255) / 256, 256, 0, stream>>>(wk, WkT);
  conv_qkv<<<(CC * CC + 255) / 256, 256, 0, stream>>>(wv, WvT);
  conv_T<<<(CC * CC + 255) / 256, 256, 0, stream>>>(w_proj, WpT, CC, CC);
  conv_T<<<(CC * FF + 255) / 256, 256, 0, stream>>>(w1, W1T, CC, FF);
  conv_T<<<(CC * FF + 255) / 256, 256, 0, stream>>>(w2, W2T, FF, CC);

  dim3 g6(CC / 64, MM / 128);    // N=384 GEMMs
  dim3 g24(FF / 64, MM / 128);   // N=1536 GEMM

  // attention sublayer
  ln_kernel<<<MM, 128, 0, stream>>>(x, ln1_g, ln1_b, h1);
  gemm_wmma<<<g6, 256, 0, stream>>>(h1, WqT, nullptr, nullptr, qf, CC, CC, 0);
  gemm_wmma<<<g6, 256, 0, stream>>>(h1, WkT, nullptr, nullptr, kf, CC, CC, 0);
  gemm_wmma<<<g6, 256, 0, stream>>>(h1, WvT, nullptr, nullptr, vTf, CC, CC, 1);
  size_t attn_lds = ((size_t)TT * TT + (size_t)HS * TT) * sizeof(_Float16); // 160 KB
  attn_kernel<<<BB * HH, 256, attn_lds, stream>>>(qf, kf, vTf, attn);
  gemm_wmma<<<g6, 256, 0, stream>>>(attn, WpT, b_proj, x, x1, CC, CC, 2);

  // MLP sublayer
  ln_kernel<<<MM, 128, 0, stream>>>(x1, ln2_g, ln2_b, h2);
  gemm_wmma<<<g24, 256, 0, stream>>>(h2, W1T, b1, nullptr, relu, FF, CC, 3);
  gemm_wmma<<<g6, 256, 0, stream>>>(relu, W2T, b2, x1, (float*)d_out, CC, FF, 2);
}
